// TypeAwareGATv2Conv_10642928959588
// MI455X (gfx1250) — compile-verified, matrix operand order checked by reading, not run on hardware
//
#include <hip/hip_runtime.h>
#include <math.h>

#define NN   100000
#define EE   800000
#define HH   4
#define CC   32
#define FIN  128
#define FOUT 128
#define FCAT 160
#define SCAN_T 1024

typedef __attribute__((ext_vector_type(16))) _Float16 v16h;
typedef __attribute__((ext_vector_type(8)))  _Float16 v8h;
typedef __attribute__((ext_vector_type(8)))  float    v8f;
typedef __attribute__((ext_vector_type(4)))  int      v4i;

#if defined(__has_builtin)
#  if __has_builtin(__builtin_amdgcn_global_load_async_to_lds_b128)
#    define HAVE_ASYNC_LDS 1
#  endif
#  if __has_builtin(__builtin_amdgcn_s_wait_asynccnt)
#    define HAVE_WAIT_ASYNC 1
#  endif
#endif

typedef v4i __attribute__((address_space(1)))* as1_v4i_p;
typedef v4i __attribute__((address_space(3)))* as3_v4i_p;

__device__ __forceinline__ void wait_asynccnt0() {
#if defined(HAVE_ASYNC_LDS)
#  if defined(HAVE_WAIT_ASYNC)
    __builtin_amdgcn_s_wait_asynccnt(0);
#  else
    asm volatile("s_wait_asynccnt 0x0" ::: "memory");
#  endif
#endif
}

// ---------------------------------------------------------------------
// WMMA fragment helpers (ISA 7.12.2 16-bit layouts):
// A fragment: lane m = lane&15; halves 0..7 hold K = (lane>>4)*8 + j,
// halves 8..15 hold K = 16 + (lane>>4)*8 + j.  B mirrors A with the
// output column in the lane index (B = W^T).
// C/D: vgpr r -> row r + 8*(lane>>4), col = lane&15.
// ---------------------------------------------------------------------
__device__ __forceinline__ v16h load_frag(const _Float16* p) {
    const v8h* vp = reinterpret_cast<const v8h*>(p);
    v8h lo = vp[0], hi = vp[2];   // +16 halves
    return __builtin_shufflevector(lo, hi, 0,1,2,3,4,5,6,7,8,9,10,11,12,13,14,15);
}

template <typename OutT>
__device__ __forceinline__ void wave_tile_gemm_h(
    const _Float16* __restrict__ A, int lda, int row0,
    const _Float16* __restrict__ W, int ldw, int col0,
    int K, const float* __restrict__ bias,
    OutT* __restrict__ Out, int ldo)
{
    const int lane = threadIdx.x & 31;
    const int half = lane >> 4;
    const int mrow = row0 + (lane & 15);
    const int ncol = col0 + (lane & 15);

    v8f acc = {};
    for (int kk = 0; kk < K; kk += 32) {
        v16h a = load_frag(A + (size_t)mrow * lda + kk + half * 8);
        v16h b = load_frag(W + (size_t)ncol * ldw + kk + half * 8);
        acc = __builtin_amdgcn_wmma_f32_16x16x32_f16(
            false, a, false, b, (short)0, acc, false, false);
    }
#pragma unroll
    for (int r = 0; r < 8; ++r) {
        int row = row0 + r + 8 * half;
        Out[(size_t)row * ldo + ncol] = (OutT)(acc[r] + bias[ncol]);
    }
}

// ---------------- f32 -> f16 conversion ----------------
__global__ void cvt_f16_kernel(const float* __restrict__ in,
                               _Float16* __restrict__ out, int n)
{
    int idx = blockIdx.x * blockDim.x + threadIdx.x;
    if (idx < n) out[idx] = (_Float16)in[idx];
}

// ---------------------------------------------------------------------
// Four node projections via WMMA.  One block (8 waves) per 16-row M-tile:
// the A-tile (16x128 f16 = 4KB) is staged into LDS once via the CDNA5
// async-to-LDS path, then each wave (one 16-col N-tile) loads its A
// fragments from LDS once and reuses them across all 4 weight matrices.
// ---------------------------------------------------------------------
__global__ void proj_gemm_kernel(const _Float16* __restrict__ Xh,
    const _Float16* Wh0, const _Float16* Wh1,
    const _Float16* Wh2, const _Float16* Wh3,
    const float* bl1, const float* br1, const float* bl2, const float* br2,
    _Float16* XL1, _Float16* XR1, _Float16* XL2, _Float16* XR2)
{
    __shared__ _Float16 tileA[16 * FIN];   // 4 KB

    const int mt   = blockIdx.x;           // M tile (0..NN/16-1)
    const int row0 = mt * 16;
    const int wv   = threadIdx.x >> 5;     // wave = N tile (0..7)
    const int lane = threadIdx.x & 31;
    const int half = lane >> 4;

    // cooperative A-tile fill: 256 threads x 16B
    {
        const int t    = threadIdx.x;
        const int row  = t >> 4;
        const int colh = (t & 15) * 8;
        const _Float16* g = Xh + (size_t)(row0 + row) * FIN + colh;
        _Float16*       l = &tileA[row * FIN + colh];
#if defined(HAVE_ASYNC_LDS)
        __builtin_amdgcn_global_load_async_to_lds_b128(
            (as1_v4i_p)g, (as3_v4i_p)l, 0, 0);
        wait_asynccnt0();
#else
        *reinterpret_cast<v8h*>(l) = *reinterpret_cast<const v8h*>(g);
#endif
    }
    __syncthreads();

    // A fragments from LDS, loaded once, reused for all 4 matrices
    v16h afr[4];
#pragma unroll
    for (int k4 = 0; k4 < 4; ++k4)
        afr[k4] = load_frag(&tileA[(lane & 15) * FIN + k4 * 32 + half * 8]);

    const _Float16* Ws[4] = {Wh0, Wh1, Wh2, Wh3};
    const float*    Bs[4] = {bl1, br1, bl2, br2};
    _Float16*       Os[4] = {XL1, XR1, XL2, XR2};
    const int col0 = wv * 16;
    const int ncol = col0 + (lane & 15);

#pragma unroll
    for (int mat = 0; mat < 4; ++mat) {
        const _Float16* W = Ws[mat];
        v8f acc = {};
#pragma unroll
        for (int k4 = 0; k4 < 4; ++k4) {
            v16h b = load_frag(W + (size_t)ncol * FIN + k4 * 32 + half * 8);
            acc = __builtin_amdgcn_wmma_f32_16x16x32_f16(
                false, afr[k4], false, b, (short)0, acc, false, false);
        }
        const float bv = Bs[mat][ncol];
        _Float16* O = Os[mat];
#pragma unroll
        for (int r = 0; r < 8; ++r) {
            int row = row0 + r + 8 * half;
            O[(size_t)row * FOUT + ncol] = (_Float16)(acc[r] + bv);
        }
    }
}

// ---------------- CSR build ----------------
__global__ void zero_deg_kernel(int* __restrict__ deg)
{
    int idx = blockIdx.x * blockDim.x + threadIdx.x;
    if (idx < NN) deg[idx] = 0;
}

__global__ void hist_kernel(const long long* __restrict__ ei, int* __restrict__ deg)
{
    int e = blockIdx.x * blockDim.x + threadIdx.x;
    if (e >= EE) return;
    atomicAdd(&deg[(int)ei[EE + e]], 1);
}

// single-block exclusive scan (LDS Hillis-Steele over per-thread chunk sums)
__global__ void scan_kernel(const int* __restrict__ deg,
                            int* __restrict__ rowstart, int* __restrict__ cursor)
{
    __shared__ int sums[SCAN_T];
    const int tid   = threadIdx.x;
    const int chunk = (NN + SCAN_T - 1) / SCAN_T;
    const int lo    = tid * chunk;
    const int hi    = (lo + chunk < NN) ? lo + chunk : NN;

    int s = 0;
    for (int i = lo; i < hi; ++i) s += deg[i];
    sums[tid] = s;
    __syncthreads();
    for (int off = 1; off < SCAN_T; off <<= 1) {
        int v = (tid >= off) ? sums[tid - off] : 0;
        __syncthreads();
        sums[tid] += v;
        __syncthreads();
    }
    int run = (tid == 0) ? 0 : sums[tid - 1];
    for (int i = lo; i < hi; ++i) {
        rowstart[i] = run; cursor[i] = run; run += deg[i];
    }
    if (tid == SCAN_T - 1) rowstart[NN] = run;
}

__global__ void scatter_kernel(const long long* __restrict__ ei,
                               int* __restrict__ cursor, int* __restrict__ adj)
{
    int e = blockIdx.x * blockDim.x + threadIdx.x;
    if (e >= EE) return;
    int d = (int)ei[EE + e];
    int pos = atomicAdd(&cursor[d], 1);
    adj[pos] = (int)ei[e];
}

// ---------------------------------------------------------------------
// One wave per destination node: scores, segment softmax and aggregation
// of BOTH GATv2 layers, fully in registers, no atomics.
// lane <-> channels [4*lane, 4*lane+4), head h = lane>>3.
// ---------------------------------------------------------------------
__global__ void node_attn_kernel(
    const int* __restrict__ rowstart, const int* __restrict__ adj,
    const _Float16* __restrict__ XL1, const _Float16* __restrict__ XR1,
    const _Float16* __restrict__ XL2, const _Float16* __restrict__ XR2,
    const float* __restrict__ att1, const float* __restrict__ att2,
    const float* __restrict__ bias1, const float* __restrict__ bias2,
    _Float16* __restrict__ FEAT)
{
    int wid = (blockIdx.x * blockDim.x + threadIdx.x) >> 5;
    if (wid >= NN) return;                   // wave-uniform
    const int lane = threadIdx.x & 31;
    const int c0 = lane * 4;

    const int beg = rowstart[wid], end = rowstart[wid + 1];

    float xr1v[4], xr2v[4], a1v[4], a2v[4];
#pragma unroll
    for (int t = 0; t < 4; ++t) {
        xr1v[t] = (float)XR1[(size_t)wid * FOUT + c0 + t];
        xr2v[t] = (float)XR2[(size_t)wid * FOUT + c0 + t];
        a1v[t]  = att1[c0 + t];
        a2v[t]  = att2[c0 + t];
    }

    // sweep 1: per-head max score
    float mx1 = -3.0e38f, mx2 = -3.0e38f;
    for (int j = beg; j < end; ++j) {
        int s = adj[j];
        if (j + 2 < end) {                    // hide L2 gather latency
            int sp = adj[j + 2];
            __builtin_prefetch(&XL1[(size_t)sp * FOUT + c0], 0, 0);
            __builtin_prefetch(&XL2[(size_t)sp * FOUT + c0], 0, 0);
        }
        float p1 = 0.f, p2 = 0.f;
#pragma unroll
        for (int t = 0; t < 4; ++t) {
            float v1 = (float)XL1[(size_t)s * FOUT + c0 + t] + xr1v[t];
            v1 = v1 > 0.f ? v1 : 0.2f * v1;
            p1 = fmaf(v1, a1v[t], p1);
            float v2 = (float)XL2[(size_t)s * FOUT + c0 + t] + xr2v[t];
            v2 = v2 > 0.f ? v2 : 0.2f * v2;
            p2 = fmaf(v2, a2v[t], p2);
        }
#pragma unroll
        for (int off = 1; off <= 4; off <<= 1) {  // reduce within 8-lane head group
            p1 += __shfl_xor(p1, off);
            p2 += __shfl_xor(p2, off);
        }
        mx1 = fmaxf(mx1, p1);
        mx2 = fmaxf(mx2, p2);
    }

    // sweep 2: exp / sum / weighted accumulate (normalize at the end)
    float sum1 = 0.f, sum2 = 0.f;
    float acc1[4] = {0.f, 0.f, 0.f, 0.f};
    float acc2[4] = {0.f, 0.f, 0.f, 0.f};
    for (int j = beg; j < end; ++j) {
        int s = adj[j];
        if (j + 2 < end) {
            int sp = adj[j + 2];
            __builtin_prefetch(&XL1[(size_t)sp * FOUT + c0], 0, 0);
            __builtin_prefetch(&XL2[(size_t)sp * FOUT + c0], 0, 0);
        }
        float l1[4], l2[4];
        float p1 = 0.f, p2 = 0.f;
#pragma unroll
        for (int t = 0; t < 4; ++t) {
            l1[t] = (float)XL1[(size_t)s * FOUT + c0 + t];
            l2[t] = (float)XL2[(size_t)s * FOUT + c0 + t];
            float v1 = l1[t] + xr1v[t];
            v1 = v1 > 0.f ? v1 : 0.2f * v1;
            p1 = fmaf(v1, a1v[t], p1);
            float v2 = l2[t] + xr2v[t];
            v2 = v2 > 0.f ? v2 : 0.2f * v2;
            p2 = fmaf(v2, a2v[t], p2);
        }
#pragma unroll
        for (int off = 1; off <= 4; off <<= 1) {
            p1 += __shfl_xor(p1, off);
            p2 += __shfl_xor(p2, off);
        }
        float w1 = __expf(p1 - mx1);
        float w2 = __expf(p2 - mx2);
        sum1 += w1; sum2 += w2;
#pragma unroll
        for (int t = 0; t < 4; ++t) {
            acc1[t] = fmaf(w1, l1[t], acc1[t]);
            acc2[t] = fmaf(w2, l2[t], acc2[t]);
        }
    }

    float r1 = (end > beg) ? 1.f / (sum1 + 1e-16f) : 0.f;
    float r2 = (end > beg) ? 1.f / (sum2 + 1e-16f) : 0.f;
#pragma unroll
    for (int t = 0; t < 4; ++t) {
        float o = acc1[t] * r1 + acc2[t] * r2 + bias1[c0 + t] + bias2[c0 + t];
        FEAT[(size_t)wid * FCAT + c0 + t] = (_Float16)o;
    }
}

// ---------------- type projection into FEAT[:,128:160] ----------------
__global__ void type_proj_kernel(const float* __restrict__ nt,
                                 const float* __restrict__ Wt,
                                 const float* __restrict__ bt,
                                 _Float16* __restrict__ FEAT)
{
    int idx = blockIdx.x * blockDim.x + threadIdx.x;   // NN*32
    if (idx >= NN * 32) return;
    int n = idx >> 5, t = idx & 31;
    float te = fmaf(nt[(size_t)n * 2 + 0], Wt[t * 2 + 0],
               fmaf(nt[(size_t)n * 2 + 1], Wt[t * 2 + 1], bt[t]));
    FEAT[(size_t)n * FCAT + FOUT + t] = (_Float16)te;
}

// ---------------- output GEMM (K=160) via WMMA, f32 out ----------------
__global__ void out_gemm_kernel(const _Float16* __restrict__ FEAT,
                                const _Float16* __restrict__ Wch,
                                const float* __restrict__ bc,
                                float* __restrict__ out)
{
    const int ntiles = FOUT / 16;
    int wave = (blockIdx.x * blockDim.x + threadIdx.x) >> 5;
    int mt = wave / ntiles, nt = wave % ntiles;
    if (mt >= NN / 16) return;               // wave-uniform
    wave_tile_gemm_h<float>(FEAT, FCAT, mt * 16, Wch, FCAT, nt * 16, FCAT, bc, out, FOUT);
}

extern "C" void kernel_launch(void* const* d_in, const int* in_sizes, int n_in,
                              void* d_out, int out_size, void* d_ws, size_t ws_size,
                              hipStream_t stream)
{
    const float*     x    = (const float*)d_in[0];
    const long long* ei   = (const long long*)d_in[1];
    const float*     ntyp = (const float*)d_in[2];
    const float *Wl1 = (const float*)d_in[3],  *bl1 = (const float*)d_in[4];
    const float *Wr1 = (const float*)d_in[5],  *br1 = (const float*)d_in[6];
    const float *att1= (const float*)d_in[7],  *bias1=(const float*)d_in[8];
    const float *Wl2 = (const float*)d_in[9],  *bl2 = (const float*)d_in[10];
    const float *Wr2 = (const float*)d_in[11], *br2 = (const float*)d_in[12];
    const float *att2= (const float*)d_in[13], *bias2=(const float*)d_in[14];
    const float *Wt  = (const float*)d_in[15], *bt  = (const float*)d_in[16];
    const float *Wc  = (const float*)d_in[17], *bc  = (const float*)d_in[18];
    float* out = (float*)d_out;
    (void)ws_size; (void)n_in; (void)in_sizes; (void)out_size;

    // workspace carve-up (256B-aligned regions)
    char* ws = (char*)d_ws;
    size_t off = 0;
    auto take = [&](size_t bytes) { void* p = ws + off; off += (bytes + 255) & ~(size_t)255; return p; };
    _Float16* Xh   = (_Float16*)take((size_t)NN * FIN * 2);
    _Float16* Wh0  = (_Float16*)take((size_t)FOUT * FIN * 2);
    _Float16* Wh1  = (_Float16*)take((size_t)FOUT * FIN * 2);
    _Float16* Wh2  = (_Float16*)take((size_t)FOUT * FIN * 2);
    _Float16* Wh3  = (_Float16*)take((size_t)FOUT * FIN * 2);
    _Float16* Wch  = (_Float16*)take((size_t)FOUT * FCAT * 2);
    _Float16* XL1  = (_Float16*)take((size_t)NN * FOUT * 2);
    _Float16* XR1  = (_Float16*)take((size_t)NN * FOUT * 2);
    _Float16* XL2  = (_Float16*)take((size_t)NN * FOUT * 2);
    _Float16* XR2  = (_Float16*)take((size_t)NN * FOUT * 2);
    _Float16* FEAT = (_Float16*)take((size_t)NN * FCAT * 2);
    int* deg      = (int*)take((size_t)NN * 4);
    int* rowstart = (int*)take((size_t)(NN + 1) * 4);
    int* cursor   = (int*)take((size_t)NN * 4);
    int* adj      = (int*)take((size_t)EE * 4);

    const int BLK = 256;

    // f32 -> f16 pre-conversions
    cvt_f16_kernel<<<(NN * FIN + BLK - 1) / BLK, BLK, 0, stream>>>(x, Xh, NN * FIN);
    cvt_f16_kernel<<<(FOUT * FIN + BLK - 1) / BLK, BLK, 0, stream>>>(Wl1, Wh0, FOUT * FIN);
    cvt_f16_kernel<<<(FOUT * FIN + BLK - 1) / BLK, BLK, 0, stream>>>(Wr1, Wh1, FOUT * FIN);
    cvt_f16_kernel<<<(FOUT * FIN + BLK - 1) / BLK, BLK, 0, stream>>>(Wl2, Wh2, FOUT * FIN);
    cvt_f16_kernel<<<(FOUT * FIN + BLK - 1) / BLK, BLK, 0, stream>>>(Wr2, Wh3, FOUT * FIN);
    cvt_f16_kernel<<<(FOUT * FCAT + BLK - 1) / BLK, BLK, 0, stream>>>(Wc, Wch, FOUT * FCAT);

    // CSR build (dst-sorted adjacency of source ids)
    zero_deg_kernel<<<(NN + BLK - 1) / BLK, BLK, 0, stream>>>(deg);
    hist_kernel<<<(EE + BLK - 1) / BLK, BLK, 0, stream>>>(ei, deg);
    scan_kernel<<<1, SCAN_T, 0, stream>>>(deg, rowstart, cursor);
    scatter_kernel<<<(EE + BLK - 1) / BLK, BLK, 0, stream>>>(ei, cursor, adj);

    // node projections: one block per 16-row M-tile, A staged in LDS
    proj_gemm_kernel<<<NN / 16, BLK, 0, stream>>>(
        Xh, Wh0, Wh1, Wh2, Wh3, bl1, br1, bl2, br2, XL1, XR1, XL2, XR2);

    // fused attention softmax + aggregation: one wave per node, no atomics
    node_attn_kernel<<<(NN * 32 + BLK - 1) / BLK, BLK, 0, stream>>>(
        rowstart, adj, XL1, XR1, XL2, XR2, att1, att2, bias1, bias2, FEAT);

    // type projection
    type_proj_kernel<<<(NN * 32 + BLK - 1) / BLK, BLK, 0, stream>>>(ntyp, Wt, bt, FEAT);

    // output GEMM: 6250 * 8 wave-tiles
    {
        int waves = (NN / 16) * (FOUT / 16);
        out_gemm_kernel<<<waves / (BLK / 32), BLK, 0, stream>>>(FEAT, Wch, bc, out);
    }
}
